// DRModel_59785944761107
// MI455X (gfx1250) — compile-verified, matrix operand order checked by reading, not run on hardware
//
#include <hip/hip_runtime.h>
#include <hip/hip_bf16.h>

// ---------------------------------------------------------------------------
// Problem constants
// ---------------------------------------------------------------------------
#define V_VOCAB 50000
#define D_DIM   128
#define B_BATCH 64
#define S_SEQ   20
#define K_ITEMS 20
#define BS      (B_BATCH * S_SEQ)      // 1280 rows of dynamic_user / logits
#define G3D     (3 * D_DIM)            // 384
#define NTILES  (V_VOCAB / 16)         // 3125 N-tiles of the big GEMM
#define NC      25                     // vocab chunks per row-block
#define TPC     (NTILES / NC)          // 125 tiles per chunk (exact)

#define NEG_BIG (-3.0e38f)

typedef __attribute__((ext_vector_type(16))) __bf16 v16bf;
typedef __attribute__((ext_vector_type(8)))  __bf16 v8bf;
typedef __attribute__((ext_vector_type(8)))  float  v8f;

// ---------------------------------------------------------------------------
// CDNA5 WMMA helpers (wave32).  A: lane<16 -> row lane, K = {0..7,16..23};
// lane>=16 -> row lane-16, K = {8..15,24..31}.  B symmetric with lane = column.
// Both load identically from a row-major [16][ld] block (ld in elements).
// ---------------------------------------------------------------------------
__device__ __forceinline__ v16bf load_tile_bf16(const __bf16* base, int ld, int lane) {
    int r  = lane & 15;
    int hi = (lane >> 4) & 1;
    const __bf16* p = base + (size_t)r * ld + hi * 8;
    v8bf lo = *(const v8bf*)(p);
    v8bf hh = *(const v8bf*)(p + 16);
    v16bf o;
#pragma unroll
    for (int i = 0; i < 8; ++i) { o[i] = lo[i]; o[i + 8] = hh[i]; }
    return o;
}

__device__ __forceinline__ v8f wmma_bf16(v16bf a, v16bf b, v8f c) {
    return __builtin_amdgcn_wmma_f32_16x16x32_bf16(false, a, false, b,
                                                   (short)0, c, false, false);
}

// C/D layout: lane 0-15: N=lane, M=vgpr i; lane 16-31: N=lane-16, M=8+i.
__device__ __forceinline__ void store_tile_f32(float* base, int ld, int lane, const v8f& c) {
    int n  = lane & 15;
    int hi = (lane >> 4) & 1;
    float* p = base + (size_t)(hi * 8) * ld + n;
#pragma unroll
    for (int i = 0; i < 8; ++i) p[(size_t)i * ld] = c[i];
}

__device__ __forceinline__ void combine_ms(float& m, float& s, float m2, float s2) {
    float mN = fmaxf(m, m2);
    s = s * __expf(m - mN) + s2 * __expf(m2 - mN);
    m = mN;
}

// ---------------------------------------------------------------------------
// K1: basket pooling -> pooled_bf16 [BS][128]
// ---------------------------------------------------------------------------
__global__ __launch_bounds__(128) void k_pool(const int* __restrict__ items,
                                              const float* __restrict__ dollars,
                                              const float* __restrict__ emb,
                                              __bf16* __restrict__ pooled) {
    int bs = blockIdx.x;            // b*S + s
    int d  = threadIdx.x;
    const int*   it = items   + (size_t)bs * K_ITEMS;
    const float* dl = dollars + (size_t)bs * K_ITEMS;
    float acc = 0.f;
#pragma unroll
    for (int k = 0; k < K_ITEMS; ++k)
        acc += emb[(size_t)it[k] * D_DIM + d] * dl[k];
    pooled[(size_t)bs * D_DIM + d] = (__bf16)(acc * (1.0f / K_ITEMS));
}

// ---------------------------------------------------------------------------
// K2: W_ih / W_hh -> bf16 (row-major [384][128])
// ---------------------------------------------------------------------------
__global__ __launch_bounds__(256) void k_wconv(const float* __restrict__ wih,
                                               const float* __restrict__ whh,
                                               __bf16* __restrict__ wihb,
                                               __bf16* __restrict__ whhb) {
    int i = blockIdx.x * 256 + threadIdx.x;
    if (i < G3D * D_DIM) {
        wihb[i] = (__bf16)wih[i];
        whhb[i] = (__bf16)whh[i];
    }
}

// ---------------------------------------------------------------------------
// K3: L1-normalize emb rows -> item_bf16 [V][128]
// ---------------------------------------------------------------------------
__global__ __launch_bounds__(128) void k_item(const float* __restrict__ emb,
                                              __bf16* __restrict__ itemb) {
    int v = blockIdx.x;
    int d = threadIdx.x;
    float x = emb[(size_t)v * D_DIM + d];
    float a = fabsf(x);
#pragma unroll
    for (int msk = 16; msk >= 1; msk >>= 1) a += __shfl_xor(a, msk, 32);
    __shared__ float part[4];
    int lane = threadIdx.x & 31, wv = threadIdx.x >> 5;
    if (lane == 0) part[wv] = a;
    __syncthreads();
    float norm = part[0] + part[1] + part[2] + part[3];
    if (norm == 0.f) norm = 1.f;
    itemb[(size_t)v * D_DIM + d] = (__bf16)(x / norm);
}

// ---------------------------------------------------------------------------
// K4: GI = pooled @ W_ih^T   [1280 x 384], WMMA, fully parallel
//     tiles: 80 x 24 = 1920, 240 blocks x 8 waves
// ---------------------------------------------------------------------------
__global__ __launch_bounds__(256) void k_gi(const __bf16* __restrict__ pooled,
                                            const __bf16* __restrict__ wihb,
                                            float* __restrict__ GI) {
    int tile = blockIdx.x * 8 + (threadIdx.x >> 5);
    if (tile >= (BS / 16) * (G3D / 16)) return;
    int lane = threadIdx.x & 31;
    int mb = tile / (G3D / 16);
    int nb = tile % (G3D / 16);
    const __bf16* ab = pooled + (size_t)mb * 16 * D_DIM;
    const __bf16* bb = wihb   + (size_t)nb * 16 * D_DIM;   // B col n = W_ih row n
    v8f acc = {};
#pragma unroll
    for (int kk = 0; kk < 4; ++kk) {
        v16bf a = load_tile_bf16(ab + kk * 32, D_DIM, lane);
        v16bf b = load_tile_bf16(bb + kk * 32, D_DIM, lane);
        acc = wmma_bf16(a, b, acc);
    }
    store_tile_f32(GI + (size_t)mb * 16 * G3D + nb * 16, G3D, lane, acc);
}

// ---------------------------------------------------------------------------
// K5: sequential GRU, single persistent block (16 waves), h in LDS
// ---------------------------------------------------------------------------
__global__ __launch_bounds__(512) void k_gru(const float* __restrict__ hidden,
                                             const float* __restrict__ GI,
                                             const __bf16* __restrict__ whhb,
                                             const float* __restrict__ b_ih,
                                             const float* __restrict__ b_hh,
                                             float* __restrict__ gh,
                                             __bf16* __restrict__ dyn) {
    __shared__ __align__(16) __bf16 hbf[B_BATCH * D_DIM];
    __shared__ float hf[B_BATCH * D_DIM];
    int lane = threadIdx.x & 31, wave = threadIdx.x >> 5;

    for (int e = threadIdx.x; e < B_BATCH * D_DIM; e += 512) {
        float h0 = hidden[e];            // hidden[0] : [B][D]
        hf[e]  = h0;
        hbf[e] = (__bf16)h0;
    }
    __syncthreads();

    for (int t = 0; t < S_SEQ; ++t) {
        // gh = h @ W_hh^T : 4 x 24 = 96 tiles over 16 waves
        for (int tile = wave; tile < (B_BATCH / 16) * (G3D / 16); tile += 16) {
            int mb = tile / (G3D / 16);
            int nb = tile % (G3D / 16);
            const __bf16* ab = hbf  + (size_t)mb * 16 * D_DIM;
            const __bf16* bb = whhb + (size_t)nb * 16 * D_DIM;
            v8f acc = {};
#pragma unroll
            for (int kk = 0; kk < 4; ++kk) {
                v16bf a = load_tile_bf16(ab + kk * 32, D_DIM, lane);
                v16bf b = load_tile_bf16(bb + kk * 32, D_DIM, lane);
                acc = wmma_bf16(a, b, acc);
            }
            store_tile_f32(gh + (size_t)mb * 16 * G3D + nb * 16, G3D, lane, acc);
        }
        __threadfence();
        __syncthreads();

        // gates: 8192 elements, 16 per thread
        float hn[16];
#pragma unroll
        for (int j = 0; j < 16; ++j) {
            int e = threadIdx.x + 512 * j;
            int b = e >> 7, d = e & 127;
            size_t gir = (size_t)(b * S_SEQ + t) * G3D;
            size_t ghr = (size_t)b * G3D;
            float gr  = GI[gir + d]       + b_ih[d]       + gh[ghr + d]       + b_hh[d];
            float gz  = GI[gir + 128 + d] + b_ih[128 + d] + gh[ghr + 128 + d] + b_hh[128 + d];
            float gin = GI[gir + 256 + d] + b_ih[256 + d];
            float ghn = gh[ghr + 256 + d] + b_hh[256 + d];
            float rg = 1.f / (1.f + __expf(-gr));
            float zg = 1.f / (1.f + __expf(-gz));
            float ng = tanhf(gin + rg * ghn);
            hn[j] = (1.f - zg) * ng + zg * hf[e];
        }
        __syncthreads();
#pragma unroll
        for (int j = 0; j < 16; ++j) {
            int e = threadIdx.x + 512 * j;
            int b = e >> 7, d = e & 127;
            hf[e]  = hn[j];
            hbf[e] = (__bf16)hn[j];
            dyn[(size_t)(b * S_SEQ + t) * D_DIM + d] = (__bf16)hn[j];
        }
        __syncthreads();
    }
}

// ---------------------------------------------------------------------------
// K6: logits = dyn @ item^T (WMMA) -> d_out raw, + per-block online softmax
//     partials (m, sumexp) per row.  grid = 80 rowblocks x NC chunks.
//     A block (16x128 bf16, 4KB) is staged once per block into LDS with
//     CDNA5 async-to-LDS loads (ASYNCcnt path) instead of 8 redundant
//     per-wave global loads.
// ---------------------------------------------------------------------------
__global__ __launch_bounds__(256) void k_logits(const __bf16* __restrict__ dyn,
                                                const __bf16* __restrict__ item,
                                                float* __restrict__ out,
                                                float2* __restrict__ partial) {
    int rb   = blockIdx.x / NC;
    int ch   = blockIdx.x % NC;
    int lane = threadIdx.x & 31;
    int wave = threadIdx.x >> 5;
    int r    = lane & 15;
    int hi   = (lane >> 4) & 1;

    // --- stage the 16x128 A block into LDS via async global->LDS copies ----
    __shared__ __align__(16) __bf16 aShared[16 * D_DIM];          // 4 KB
    if (wave == 0) {
        unsigned ldsBase = (unsigned)(uintptr_t)(void*)aShared;   // LDS offset = addr[31:0]
        const char* g = (const char*)(dyn + (size_t)rb * 16 * D_DIM);
#pragma unroll
        for (int it = 0; it < 8; ++it) {                          // 8 * 32 lanes * 16B = 4KB
            unsigned           ldsOff = ldsBase + (unsigned)((it * 32 + lane) * 16);
            unsigned long long gaddr  = (unsigned long long)(uintptr_t)(g + (it * 32 + lane) * 16);
            asm volatile("global_load_async_to_lds_b128 %0, %1, off"
                         :: "v"(ldsOff), "v"(gaddr)
                         : "memory");
        }
        asm volatile("s_wait_asynccnt 0x0" ::: "memory");         // ASYNCcnt == 0
    }
    __syncthreads();

    // A block: 16 rows x 128 K, kept in registers across the whole chunk
    v16bf a[4];
#pragma unroll
    for (int kk = 0; kk < 4; ++kk) a[kk] = load_tile_bf16(aShared + kk * 32, D_DIM, lane);

    float mrun[8], srun[8];
#pragma unroll
    for (int i = 0; i < 8; ++i) { mrun[i] = NEG_BIG; srun[i] = 0.f; }

    int t0 = ch * TPC;
    for (int t = t0 + wave; t < t0 + TPC; t += 8) {
        const __bf16* bb = item + (size_t)t * 16 * D_DIM;
        __builtin_prefetch(bb + (size_t)8 * 16 * D_DIM, 0, 1);  // next tile for this wave
        v8f acc = {};
#pragma unroll
        for (int kk = 0; kk < 4; ++kk) {
            v16bf b = load_tile_bf16(bb + kk * 32, D_DIM, lane);
            acc = wmma_bf16(a[kk], b, acc);
        }
        float* ob = out + (size_t)(rb * 16 + hi * 8) * V_VOCAB + (size_t)t * 16 + r;
#pragma unroll
        for (int i = 0; i < 8; ++i) {
            float x = acc[i];
            ob[(size_t)i * V_VOCAB] = x;
            float mN = fmaxf(mrun[i], x);
            srun[i] = srun[i] * __expf(mrun[i] - mN) + __expf(x - mN);
            mrun[i] = mN;
        }
    }

    // reduce (m,s) across the 16 lanes of each half-wave (rows live per half)
#pragma unroll
    for (int i = 0; i < 8; ++i) {
        float m = mrun[i], s = srun[i];
        for (int msk = 1; msk < 16; msk <<= 1) {
            float m2 = __shfl_xor(m, msk, 32);
            float s2 = __shfl_xor(s, msk, 32);
            combine_ms(m, s, m2, s2);
        }
        mrun[i] = m; srun[i] = s;
    }

    __shared__ float sm[16][8], ss[16][8];
    if (r == 0) {
#pragma unroll
        for (int i = 0; i < 8; ++i) {
            sm[hi * 8 + i][wave] = mrun[i];
            ss[hi * 8 + i][wave] = srun[i];
        }
    }
    __syncthreads();
    if (threadIdx.x < 16) {
        float m = NEG_BIG, s = 0.f;
        for (int w = 0; w < 8; ++w) combine_ms(m, s, sm[threadIdx.x][w], ss[threadIdx.x][w]);
        partial[(size_t)blockIdx.x * 16 + threadIdx.x] = make_float2(m, s);
    }
}

// ---------------------------------------------------------------------------
// K7: reduce NC partials per row -> final (m, s)
// ---------------------------------------------------------------------------
__global__ __launch_bounds__(256) void k_reduce(const float2* __restrict__ partial,
                                                float2* __restrict__ ms) {
    int rr = blockIdx.x * 256 + threadIdx.x;
    if (rr >= BS) return;
    int rb = rr >> 4, r16 = rr & 15;
    float m = NEG_BIG, s = 0.f;
    for (int c = 0; c < NC; ++c) {
        float2 p = partial[(size_t)(rb * NC + c) * 16 + r16];
        combine_ms(m, s, p.x, p.y);
    }
    ms[rr] = make_float2(m, s);
}

// ---------------------------------------------------------------------------
// K8: normalize in place: out = exp(x - m) / s   (float4 vectorized)
// ---------------------------------------------------------------------------
__global__ __launch_bounds__(256) void k_norm(float* __restrict__ out,
                                              const float2* __restrict__ ms) {
    size_t i4  = (size_t)blockIdx.x * 256 + threadIdx.x;   // 16,000,000 total
    int    row = (int)(i4 / (V_VOCAB / 4));
    float2 p   = ms[row];
    float  inv = 1.0f / p.y;                               // s >= 1 always
    float4* o = (float4*)out;
    float4 v = o[i4];
    v.x = __expf(v.x - p.x) * inv;
    v.y = __expf(v.y - p.x) * inv;
    v.z = __expf(v.z - p.x) * inv;
    v.w = __expf(v.w - p.x) * inv;
    o[i4] = v;
}

// ---------------------------------------------------------------------------
// Launch
// ---------------------------------------------------------------------------
extern "C" void kernel_launch(void* const* d_in, const int* in_sizes, int n_in,
                              void* d_out, int out_size, void* d_ws, size_t ws_size,
                              hipStream_t stream) {
    const int*   items   = (const int*)  d_in[0];
    const float* dollars = (const float*)d_in[1];
    const float* hidden  = (const float*)d_in[2];
    const float* emb     = (const float*)d_in[3];
    const float* W_ih    = (const float*)d_in[4];
    const float* W_hh    = (const float*)d_in[5];
    const float* b_ih    = (const float*)d_in[6];
    const float* b_hh    = (const float*)d_in[7];
    float* out = (float*)d_out;

    char* ws = (char*)d_ws;
    // workspace layout (bytes, 256-aligned)
    __bf16* pooled_bf = (__bf16*)(ws + 0);          //   327,680
    __bf16* wih_bf    = (__bf16*)(ws + 327680);     //    98,304
    __bf16* whh_bf    = (__bf16*)(ws + 425984);     //    98,304
    __bf16* item_bf   = (__bf16*)(ws + 524288);     // 12,800,000
    float*  GI        = (float*) (ws + 13324288);   //  1,966,080
    float*  gh        = (float*) (ws + 15290368);   //     98,304
    __bf16* dyn_bf    = (__bf16*)(ws + 15388672);   //    327,680
    float2* partial   = (float2*)(ws + 15716352);   //    256,000
    float2* msbuf     = (float2*)(ws + 16044032);   //     10,240

    k_pool  <<<BS,                128, 0, stream>>>(items, dollars, emb, pooled_bf);
    k_wconv <<<(G3D * D_DIM + 255) / 256, 256, 0, stream>>>(W_ih, W_hh, wih_bf, whh_bf);
    k_item  <<<V_VOCAB,           128, 0, stream>>>(emb, item_bf);
    k_gi    <<<(BS / 16) * (G3D / 16) / 8, 256, 0, stream>>>(pooled_bf, wih_bf, GI);
    k_gru   <<<1,                 512, 0, stream>>>(hidden, GI, whh_bf, b_ih, b_hh, gh, dyn_bf);
    k_logits<<<(BS / 16) * NC,    256, 0, stream>>>(dyn_bf, item_bf, out, partial);
    k_reduce<<<(BS + 255) / 256,  256, 0, stream>>>(partial, msbuf);
    k_norm  <<<(BS * V_VOCAB / 4) / 256, 256, 0, stream>>>(out, msbuf);
}
